// GAT_82695300317417
// MI455X (gfx1250) — matrix operand order repaired and checked by reading, not compiled, not run on hardware
//
#include <hip/hip_runtime.h>
#include <math.h>

typedef __attribute__((ext_vector_type(16))) __bf16 v16bf;
typedef __attribute__((ext_vector_type(8)))  float  v8f;

// ---------------------------------------------------------------------------
// Float atomic max via order-preserving int trick (monotone under both ops).
// ---------------------------------------------------------------------------
__device__ inline void atomicMaxF(float* addr, float val) {
  int iv = __float_as_int(val);
  if (iv >= 0) {
    atomicMax((int*)addr, iv);
  } else {
    atomicMin((unsigned int*)addr, (unsigned int)iv);
  }
}

// ---------------------------------------------------------------------------
// Z = A[M,K] @ W[K,128]  via v_wmma_f32_16x16x32_bf16.
// Block = 128 threads (4 waves). Block owns 16 rows of A and all 128 cols.
//
// LDS holds the tiles in FRAGMENT-MAJOR order: for each 16x16x32 WMMA step,
// each lane's 16 bf16 operand values are 32 contiguous (32B-aligned) bytes,
// so a fragment load is two ds_load_b128 instead of 16 ds_load_u16.
//
// Forward WMMA operand layouts (CDNA5 ISA 7.12.2), k = K-within-step (0..31):
//   A: lane = (k[3])*16 + m,  slot = (k[4]<<3) | k[2:0]
//   B: lane = (k[3])*16 + n,  slot = (k[4]<<3) | k[2:0]
//   C/D 16x16 f32: lane n = L&15; VGPR r holds M = r + 8*(L>>4)
//
// Boundary handling is a block-uniform branch: interior blocks (the common
// case; all blocks when M % 16 == 0) run fully unguarded straight-line code.
// ---------------------------------------------------------------------------
template <int K>
__global__ __launch_bounds__(128) void gat_gemm_wmma(const float* __restrict__ A,
                                                     const float* __restrict__ W,
                                                     float* __restrict__ Z, int M) {
  constexpr int KSTEPS = K / 32;
  // sAf[ks][lane][slot], sBf[ks][nsub][lane][slot]; slot = 16 bf16 = 32 bytes
  __shared__ __align__(16) __bf16 sAf[KSTEPS * 32 * 16];
  __shared__ __align__(16) __bf16 sBf[KSTEPS * 8 * 32 * 16];

  const int tid   = threadIdx.x;
  const int row0  = blockIdx.x * 16;
  const bool full = (row0 + 16 <= M);

  // Stage A tile (16 x K), coalesced global read, fragment-major LDS write.
  if (full) {
#pragma unroll
    for (int i = tid; i < 16 * K; i += 128) {
      int r = i / K, c = i % K;
      float v = A[(size_t)(row0 + r) * K + c];
      int ks = c >> 5, kr = c & 31;
      int lane = ((kr >> 3) & 1) * 16 + r;
      int slot = ((kr >> 4) << 3) | (kr & 7);
      sAf[((ks * 32 + lane) << 4) + slot] = (__bf16)v;
    }
  } else {
#pragma unroll
    for (int i = tid; i < 16 * K; i += 128) {
      int r = i / K, c = i % K;
      int gr = row0 + r;
      float v = (gr < M) ? A[(size_t)gr * K + c] : 0.0f;
      int ks = c >> 5, kr = c & 31;
      int lane = ((kr >> 3) & 1) * 16 + r;
      int slot = ((kr >> 4) << 3) | (kr & 7);
      sAf[((ks * 32 + lane) << 4) + slot] = (__bf16)v;
    }
  }
  // Stage W (K x 128), coalesced global read, fragment-major LDS write.
#pragma unroll 4
  for (int i = tid; i < K * 128; i += 128) {
    int k = i >> 7, n = i & 127;
    int ks = k >> 5, kr = k & 31;
    int lane = ((kr >> 3) & 1) * 16 + (n & 15);
    int slot = ((kr >> 4) << 3) | (kr & 7);
    int nsub = n >> 4;
    sBf[(((ks * 8 + nsub) * 32 + lane) << 4) + slot] = (__bf16)W[i];
  }
  __syncthreads();

  const int wave = tid >> 5;
  const int lane = tid & 31;

  v8f acc0 = {}; v8f acc1 = {};
#pragma unroll
  for (int ks = 0; ks < KSTEPS; ++ks) {
    const v16bf af  = *(const v16bf*)&sAf[(ks * 32 + lane) << 4];
    const v16bf bf0 = *(const v16bf*)&sBf[((ks * 8 + wave * 2 + 0) * 32 + lane) << 4];
    const v16bf bf1 = *(const v16bf*)&sBf[((ks * 8 + wave * 2 + 1) * 32 + lane) << 4];
    acc0 = __builtin_amdgcn_wmma_f32_16x16x32_bf16(false, af, false, bf0,
                                                   (short)0, acc0, false, false);
    acc1 = __builtin_amdgcn_wmma_f32_16x16x32_bf16(false, af, false, bf1,
                                                   (short)0, acc1, false, false);
  }

  // Epilogue: lane stores column (lane&15) of rows r / r+8; acc1's block is
  // 16 floats to the right of acc0's -> pairable 64B-offset stores.
  float* zbase = Z + (size_t)(row0 + 8 * (lane >> 4)) * 128 + wave * 32 + (lane & 15);
  if (full) {
#pragma unroll
    for (int r = 0; r < 8; ++r) {
      zbase[(size_t)r * 128]      = acc0[r];
      zbase[(size_t)r * 128 + 16] = acc1[r];
    }
  } else {
#pragma unroll
    for (int r = 0; r < 8; ++r) {
      int gr = row0 + r + 8 * (lane >> 4);
      if (gr < M) {
        zbase[(size_t)r * 128]      = acc0[r];
        zbase[(size_t)r * 128 + 16] = acc1[r];
      }
    }
  }
}

// ---------------------------------------------------------------------------
// el[n,h] = <z[n,h,:], a_l[h,:]>, er likewise. Wave per node (wave32):
// lane covers 4 dims; lanes 0..15 = head 0, 16..31 = head 1; 16-lane reduce.
// ---------------------------------------------------------------------------
__global__ __launch_bounds__(256) void gat_scores(const float* __restrict__ Z,
                                                  const float* __restrict__ al,
                                                  const float* __restrict__ ar,
                                                  float* __restrict__ el,
                                                  float* __restrict__ er, int N) {
  int node = blockIdx.x * 8 + (threadIdx.x >> 5);
  int lane = threadIdx.x & 31;
  if (node >= N) return;
  const float4 z = *(const float4*)(Z + (size_t)node * 128 + lane * 4);
  const float4 l = *(const float4*)(al + lane * 4);
  const float4 r = *(const float4*)(ar + lane * 4);
  float pl = z.x * l.x + z.y * l.y + z.z * l.z + z.w * l.w;
  float pr = z.x * r.x + z.y * r.y + z.z * r.z + z.w * r.w;
#pragma unroll
  for (int off = 8; off >= 1; off >>= 1) {
    pl += __shfl_xor(pl, off, 32);
    pr += __shfl_xor(pr, off, 32);
  }
  if ((lane & 15) == 0) {
    int h = lane >> 4;
    el[node * 2 + h] = pl;
    er[node * 2 + h] = pr;
  }
}

__global__ void gat_fill(float* __restrict__ p, float v, int n) {
  int i = blockIdx.x * blockDim.x + threadIdx.x;
  if (i < n) p[i] = v;
}

// e = leaky_relu(el[src]+er[dst]); segment max into emax[dst].
__global__ __launch_bounds__(256) void gat_edge_max(const int* __restrict__ src,
                                                    const int* __restrict__ dst,
                                                    const float* __restrict__ el,
                                                    const float* __restrict__ er,
                                                    float* __restrict__ esc,
                                                    float* __restrict__ emax, int E) {
  int e = blockIdx.x * blockDim.x + threadIdx.x;
  if (e >= E) return;
  int s = src[e], d = dst[e];
#pragma unroll
  for (int h = 0; h < 2; ++h) {
    float v = el[s * 2 + h] + er[d * 2 + h];
    v = v > 0.0f ? v : 0.2f * v;
    esc[(size_t)e * 2 + h] = v;
    atomicMaxF(&emax[d * 2 + h], v);
  }
}

// ee = exp(e - emax[dst]); denom[dst] += ee (stored in place of scores).
__global__ __launch_bounds__(256) void gat_edge_exp(const int* __restrict__ dst,
                                                    const float* __restrict__ emax,
                                                    float* __restrict__ esc,
                                                    float* __restrict__ denom, int E) {
  int e = blockIdx.x * blockDim.x + threadIdx.x;
  if (e >= E) return;
  int d = dst[e];
#pragma unroll
  for (int h = 0; h < 2; ++h) {
    float ee = __expf(esc[(size_t)e * 2 + h] - emax[d * 2 + h]);
    esc[(size_t)e * 2 + h] = ee;
    atomicAdd(&denom[d * 2 + h], ee);
  }
}

// acc[dst] += ee * z[src]  (unnormalized; divide by denom at node level).
// Wave per edge; lane covers 4 dims (float4 gather, 4 atomic f32 adds).
__global__ __launch_bounds__(256) void gat_edge_aggr(const int* __restrict__ src,
                                                     const int* __restrict__ dst,
                                                     const float* __restrict__ esc,
                                                     const float* __restrict__ Z,
                                                     float* __restrict__ acc, int E) {
  int e = blockIdx.x * 8 + (threadIdx.x >> 5);
  int lane = threadIdx.x & 31;
  if (e >= E) return;
  int s = src[e], d = dst[e];
  float w = esc[(size_t)e * 2 + (lane >> 4)];
  const float4 z = *(const float4*)(Z + (size_t)s * 128 + lane * 4);
  float* out = acc + (size_t)d * 128 + lane * 4;
  atomicAdd(out + 0, w * z.x);
  atomicAdd(out + 1, w * z.y);
  atomicAdd(out + 2, w * z.z);
  atomicAdd(out + 3, w * z.w);
}

// Finalize layer 0: out = acc/denom + b, mean heads, LayerNorm(64), ReLU.
__global__ __launch_bounds__(256) void gat_final_ln(const float* __restrict__ acc,
                                                    const float* __restrict__ denom,
                                                    const float* __restrict__ b,
                                                    const float* __restrict__ gamma,
                                                    const float* __restrict__ beta,
                                                    float* __restrict__ hout, int N) {
  int node = blockIdx.x * 8 + (threadIdx.x >> 5);
  int lane = threadIdx.x & 31;
  if (node >= N) return;
  float d0 = denom[node * 2 + 0], d1 = denom[node * 2 + 1];
  float i0 = d0 > 0.0f ? 1.0f / d0 : 0.0f;
  float i1 = d1 > 0.0f ? 1.0f / d1 : 0.0f;
  float v[2];
#pragma unroll
  for (int j = 0; j < 2; ++j) {
    int d = lane * 2 + j;
    float h0 = acc[(size_t)node * 128 + d] * i0 + b[d];
    float h1 = acc[(size_t)node * 128 + 64 + d] * i1 + b[64 + d];
    v[j] = 0.5f * (h0 + h1);
  }
  float s = v[0] + v[1];
  float s2 = v[0] * v[0] + v[1] * v[1];
#pragma unroll
  for (int off = 16; off >= 1; off >>= 1) {
    s  += __shfl_xor(s, off, 32);
    s2 += __shfl_xor(s2, off, 32);
  }
  float mu  = s * (1.0f / 64.0f);
  float var = s2 * (1.0f / 64.0f) - mu * mu;
  float rstd = rsqrtf(var + 1e-5f);
#pragma unroll
  for (int j = 0; j < 2; ++j) {
    int d = lane * 2 + j;
    float o = (v[j] - mu) * rstd * gamma[d] + beta[d];
    hout[(size_t)node * 64 + d] = o > 0.0f ? o : 0.0f;
  }
}

// Finalize layer 1: out = mean_h(acc/denom + b)  (no norm / activation).
__global__ __launch_bounds__(256) void gat_final(const float* __restrict__ acc,
                                                 const float* __restrict__ denom,
                                                 const float* __restrict__ b,
                                                 float* __restrict__ out, int N) {
  int node = blockIdx.x * 8 + (threadIdx.x >> 5);
  int lane = threadIdx.x & 31;
  if (node >= N) return;
  float d0 = denom[node * 2 + 0], d1 = denom[node * 2 + 1];
  float i0 = d0 > 0.0f ? 1.0f / d0 : 0.0f;
  float i1 = d1 > 0.0f ? 1.0f / d1 : 0.0f;
#pragma unroll
  for (int j = 0; j < 2; ++j) {
    int d = lane * 2 + j;
    float h0 = acc[(size_t)node * 128 + d] * i0 + b[d];
    float h1 = acc[(size_t)node * 128 + 64 + d] * i1 + b[64 + d];
    out[(size_t)node * 64 + d] = 0.5f * (h0 + h1);
  }
}

// ---------------------------------------------------------------------------
extern "C" void kernel_launch(void* const* d_in, const int* in_sizes, int n_in,
                              void* d_out, int out_size, void* d_ws, size_t ws_size,
                              hipStream_t stream) {
  const float* feat   = (const float*)d_in[0];
  const int*   src    = (const int*)d_in[1];
  const int*   dst    = (const int*)d_in[2];
  const float* W1     = (const float*)d_in[3];
  const float* al1    = (const float*)d_in[4];
  const float* ar1    = (const float*)d_in[5];
  const float* b1     = (const float*)d_in[6];
  const float* gamma1 = (const float*)d_in[7];
  const float* beta1  = (const float*)d_in[8];
  const float* W2     = (const float*)d_in[9];
  const float* al2    = (const float*)d_in[10];
  const float* ar2    = (const float*)d_in[11];
  const float* b2     = (const float*)d_in[12];

  const int N = in_sizes[0] / 128;
  const int E = in_sizes[1];

  float* ws = (float*)d_ws;
  size_t off = 0;
  float* z     = ws + off; off += (size_t)N * 128;   // projections (reused both layers)
  float* acc   = ws + off; off += (size_t)N * 128;   // unnormalized aggregation
  float* h     = ws + off; off += (size_t)N * 64;    // layer-0 output
  float* el    = ws + off; off += (size_t)N * 2;
  float* er    = ws + off; off += (size_t)N * 2;
  float* emax  = ws + off; off += (size_t)N * 2;
  float* denom = ws + off; off += (size_t)N * 2;
  float* esc   = ws + off; off += (size_t)E * 2;

  const int nodeBlocks = (N + 7) / 8;
  const int edgeBlocks = (E + 255) / 256;
  const int aggrBlocks = (E + 7) / 8;
  const int gemmBlocks = (N + 15) / 16;
  const int fillBlocks = (N * 2 + 255) / 256;

  // ---- layer 0 ----
  hipMemsetAsync(acc, 0, (size_t)N * 128 * sizeof(float), stream);
  hipMemsetAsync(denom, 0, (size_t)N * 2 * sizeof(float), stream);
  gat_fill<<<fillBlocks, 256, 0, stream>>>(emax, -INFINITY, N * 2);

  gat_gemm_wmma<128><<<gemmBlocks, 128, 0, stream>>>(feat, W1, z, N);
  gat_scores<<<nodeBlocks, 256, 0, stream>>>(z, al1, ar1, el, er, N);
  gat_edge_max<<<edgeBlocks, 256, 0, stream>>>(src, dst, el, er, esc, emax, E);
  gat_edge_exp<<<edgeBlocks, 256, 0, stream>>>(dst, emax, esc, denom, E);
  gat_edge_aggr<<<aggrBlocks, 256, 0, stream>>>(src, dst, esc, z, acc, E);
  gat_final_ln<<<nodeBlocks, 256, 0, stream>>>(acc, denom, b1, gamma1, beta1, h, N);

  // ---- layer 1 ----
  hipMemsetAsync(acc, 0, (size_t)N * 128 * sizeof(float), stream);
  hipMemsetAsync(denom, 0, (size_t)N * 2 * sizeof(float), stream);
  gat_fill<<<fillBlocks, 256, 0, stream>>>(emax, -INFINITY, N * 2);

  gat_gemm_wmma<64><<<gemmBlocks, 128, 0, stream>>>(h, W2, z, N);
  gat_scores<<<nodeBlocks, 256, 0, stream>>>(z, al2, ar2, el, er, N);
  gat_edge_max<<<edgeBlocks, 256, 0, stream>>>(src, dst, el, er, esc, emax, E);
  gat_edge_exp<<<edgeBlocks, 256, 0, stream>>>(dst, emax, esc, denom, E);
  gat_edge_aggr<<<aggrBlocks, 256, 0, stream>>>(src, dst, esc, z, acc, E);
  gat_final<<<nodeBlocks, 256, 0, stream>>>(acc, denom, b2, (float*)d_out, N);
}
